// Attention2D_32624571580926
// MI455X (gfx1250) — matrix operand order repaired and checked by reading, not compiled, not run on hardware
//
#include <hip/hip_runtime.h>

// ---------------------------------------------------------------------------
// CDNA5 (gfx1250) wave32 WMMA attention block.
//   x:(16,512,32,32) f32 ; qkv_w:(1536,512) ; qkv_b:(1536) ; proj_w:(512,512);
//   proj_b:(512) ; out = x + proj(attn(qkv(x)))
// All matmuls via v_wmma_f32_16x16x32_bf16 (f32 accumulate). Weights are
// pre-converted to bf16 so GEMM fragment loads are pure global_load_b128.
// ---------------------------------------------------------------------------

typedef __attribute__((ext_vector_type(16))) __bf16 v16bf;
typedef __attribute__((ext_vector_type(8)))  float  v8f;
typedef __attribute__((ext_vector_type(8)))  unsigned int v8u;

#define BATCH  16
#define CCH    512
#define NTOK   1024
#define HEADS  8
#define HD     64
#define QSCALE 0.044194173824159216f   // 512^-0.5

__device__ __forceinline__ unsigned short bf16u(float f) {
  return __builtin_bit_cast(unsigned short, (__bf16)f);
}
__device__ __forceinline__ unsigned pack2(float lo, float hi) {
  return ((unsigned)bf16u(hi) << 16) | (unsigned)bf16u(lo);
}
__device__ __forceinline__ unsigned pack2u(unsigned short lo, unsigned short hi) {
  return ((unsigned)hi << 16) | (unsigned)lo;
}
__device__ __forceinline__ v16bf wmma_cast(v8u u) {
  return __builtin_bit_cast(v16bf, u);
}

// A-fragment (16x32 bf16, MxK), memory row-major [m][k], row stride rs (elems).
// lanes 0-15 -> M=lane, lanes 16-31 -> M=lane-16.
// VGPR i: K = half*16 + 2i.
__device__ __forceinline__ v16bf load_a_bf16(const unsigned short* base, int rs) {
  int lane = threadIdx.x & 31, lh = lane >> 4, m = lane & 15;
  const unsigned* row = (const unsigned*)(base + m * rs);
  v8u u;
#pragma unroll
  for (int i = 0; i < 8; ++i) {
    int kb = lh * 16 + 2 * i;
    u[i] = row[kb >> 1];
  }
  return wmma_cast(u);
}

// B-fragment (32x16 bf16, KxN), memory stored n-major [n][k], row stride rs.
// lane -> column N=lane&15; lanes 0-15 hold K=0..15, lanes 16-31 K=16..31.
__device__ __forceinline__ v16bf load_b_bf16(const unsigned short* base, int rs) {
  int lane = threadIdx.x & 31, lh = lane >> 4, n = lane & 15;
  const unsigned* row = (const unsigned*)(base + n * rs);
  v8u u;
#pragma unroll
  for (int j = 0; j < 8; ++j) u[j] = row[lh * 8 + j];
  return wmma_cast(u);
}

// B-fragment from LDS (P matrix, [n][m] layout, 32 bf16 = 16 dwords per row)
__device__ __forceinline__ v16bf load_b_lds(const unsigned* P) {
  int lane = threadIdx.x & 31, lh = lane >> 4, n = lane & 15;
  const unsigned* row = P + n * 16 + lh * 8;
  v8u u;
#pragma unroll
  for (int j = 0; j < 8; ++j) u[j] = row[j];
  return wmma_cast(u);
}

__device__ __forceinline__ v8f wmma_bf16(v16bf a, v16bf b, v8f c) {
  return __builtin_amdgcn_wmma_f32_16x16x32_bf16(false, a, false, b,
                                                 (short)0, c, false, false);
}

// ---------------------------------------------------------------------------
// Kernel 0: generic f32 -> bf16 convert (weights), 4 elems/thread.
// ---------------------------------------------------------------------------
__global__ void k_cvt(const float* __restrict__ src, unsigned short* __restrict__ dst) {
  long t = (long)blockIdx.x * 256 + threadIdx.x;
  const float* s = src + t * 4;
  uint2 v; v.x = pack2(s[0], s[1]); v.y = pack2(s[2], s[3]);
  *reinterpret_cast<uint2*>(dst + t * 4) = v;
}

// ---------------------------------------------------------------------------
// Kernel 1: x (B,C,N) f32 -> xT (B,N,C) bf16, LDS-tiled 64c x 64n transpose.
// Coalesced f32 reads, coalesced uint4 bf16 writes.
// ---------------------------------------------------------------------------
__global__ void k_xpose(const float* __restrict__ x, unsigned short* __restrict__ xT) {
  __shared__ unsigned short tile[64][66];   // +2 pad: avoid bank conflicts
  int bid = blockIdx.x;                     // 16b * 8ct * 16nt = 2048 blocks
  int ntile = bid & 15;
  int ctile = (bid >> 4) & 7;
  int b     = bid >> 7;
  int c0 = ctile * 64, n0 = ntile * 64;
  int t = threadIdx.x;

  // read: 4 threads cover one 64-wide row of n (16 consecutive f32 each)
  int cr = t >> 2, nr = (t & 3) * 16;
  const float* xp = x + ((long)b * CCH + c0 + cr) * NTOK + n0 + nr;
#pragma unroll
  for (int i = 0; i < 16; ++i) tile[cr][nr + i] = bf16u(xp[i]);
  __syncthreads();

  // write: 4 threads cover one 64-wide row of c (16 consecutive bf16 each)
  int nw = t >> 2, cw = (t & 3) * 16;
  unsigned short* op = xT + ((long)b * NTOK + n0 + nw) * CCH + c0 + cw;
  uint4 w0, w1;
  w0.x = pack2u(tile[cw + 0][nw],  tile[cw + 1][nw]);
  w0.y = pack2u(tile[cw + 2][nw],  tile[cw + 3][nw]);
  w0.z = pack2u(tile[cw + 4][nw],  tile[cw + 5][nw]);
  w0.w = pack2u(tile[cw + 6][nw],  tile[cw + 7][nw]);
  w1.x = pack2u(tile[cw + 8][nw],  tile[cw + 9][nw]);
  w1.y = pack2u(tile[cw + 10][nw], tile[cw + 11][nw]);
  w1.z = pack2u(tile[cw + 12][nw], tile[cw + 13][nw]);
  w1.w = pack2u(tile[cw + 14][nw], tile[cw + 15][nw]);
  *reinterpret_cast<uint4*>(op)     = w0;
  *reinterpret_cast<uint4*>(op + 8) = w1;
}

// ---------------------------------------------------------------------------
// Kernel 2: QKV GEMM (1536x512 @ 512xN per batch), bf16 WMMA, 32x32 wave tile.
// Writes Q,K -> [bh][n][64] bf16 (scale folded into Q), V -> [bh][64][n] bf16.
// ---------------------------------------------------------------------------
__global__ void k_qkv(const unsigned short* __restrict__ w,   // bf16 (1536,512)
                      const float* __restrict__ bias,
                      const unsigned short* __restrict__ xT,
                      unsigned short* __restrict__ Qw,
                      unsigned short* __restrict__ Kw,
                      unsigned short* __restrict__ Vw) {
  int wid = blockIdx.x * (blockDim.x >> 5) + (threadIdx.x >> 5);
  int nt = wid & 31;            // 32 n-tiles of 32
  int ot = (wid >> 5) % 48;     // 48 o-tiles of 32
  int b  = wid / (32 * 48);
  int o0 = ot * 32, n0 = nt * 32;

  v8f acc[2][2] = {};
  const unsigned short* Bbase = xT + ((long)b * NTOK + n0) * CCH;
#pragma unroll 4
  for (int kb = 0; kb < CCH; kb += 32) {
    v16bf a0 = load_a_bf16(w + (long)o0 * CCH + kb, CCH);
    v16bf a1 = load_a_bf16(w + (long)(o0 + 16) * CCH + kb, CCH);
    v16bf b0 = load_b_bf16(Bbase + kb, CCH);
    v16bf b1 = load_b_bf16(Bbase + 16 * CCH + kb, CCH);
    acc[0][0] = wmma_bf16(a0, b0, acc[0][0]);
    acc[0][1] = wmma_bf16(a0, b1, acc[0][1]);
    acc[1][0] = wmma_bf16(a1, b0, acc[1][0]);
    acc[1][1] = wmma_bf16(a1, b1, acc[1][1]);
  }

  int lane = threadIdx.x & 31, lh = lane >> 4, nn = lane & 15;
#pragma unroll
  for (int i = 0; i < 2; ++i) {
    int orow0 = o0 + 16 * i + 8 * lh;          // +r gives the output row
#pragma unroll
    for (int j = 0; j < 2; ++j) {
      int n = n0 + 16 * j + nn;
      float vals[8];
#pragma unroll
      for (int r = 0; r < 8; ++r) vals[r] = acc[i][j][r] + bias[orow0 + r];

      if (o0 < 512) {                           // ---- Q (scale folded) ----
        int h = orow0 >> 6, d0 = orow0 & 63;
        uint4 pk;
        pk.x = pack2(vals[0] * QSCALE, vals[1] * QSCALE);
        pk.y = pack2(vals[2] * QSCALE, vals[3] * QSCALE);
        pk.z = pack2(vals[4] * QSCALE, vals[5] * QSCALE);
        pk.w = pack2(vals[6] * QSCALE, vals[7] * QSCALE);
        *reinterpret_cast<uint4*>(Qw + (((long)(b * HEADS + h) * NTOK + n) * HD + d0)) = pk;
      } else if (o0 < 1024) {                   // ---- K ----
        int oq = orow0 - 512, h = oq >> 6, d0 = oq & 63;
        uint4 pk;
        pk.x = pack2(vals[0], vals[1]); pk.y = pack2(vals[2], vals[3]);
        pk.z = pack2(vals[4], vals[5]); pk.w = pack2(vals[6], vals[7]);
        *reinterpret_cast<uint4*>(Kw + (((long)(b * HEADS + h) * NTOK + n) * HD + d0)) = pk;
      } else {                                  // ---- V ([bh][d][n]) ----
        int oq = orow0 - 1024, h = oq >> 6, d0 = oq & 63;
        unsigned short* vb = Vw + ((long)(b * HEADS + h) * HD + d0) * NTOK + n;
#pragma unroll
        for (int r = 0; r < 8; ++r) vb[(long)r * NTOK] = bf16u(vals[r]);
      }
    }
  }
}

// ---------------------------------------------------------------------------
// Kernel 3: flash attention. One wave = one (b,h, 16-col n-tile); streams m in
// chunks of 32: S^T = K*Q via WMMA, online softmax per column, P staged in LDS
// as bf16 [n][m], O += V*P via WMMA. Output Ow = (B,N,C) bf16 (n-major).
// ---------------------------------------------------------------------------
__global__ void k_attn(const unsigned short* __restrict__ Qw,
                       const unsigned short* __restrict__ Kw,
                       const unsigned short* __restrict__ Vw,
                       unsigned short* __restrict__ Ow) {
  __shared__ __align__(16) unsigned ldsP[4 * 256];   // 4 waves * 16n*32m bf16
  int wslot = threadIdx.x >> 5;
  unsigned* P = ldsP + wslot * 256;

  int wid = blockIdx.x * (blockDim.x >> 5) + wslot;  // [0, 16*8*64)
  int nt = wid & 63;
  int h  = (wid >> 6) & 7;
  int b  = wid >> 9;
  int n0 = nt * 16;
  long bh = (long)(b * HEADS + h);

  const unsigned short* Qb = Qw + (bh * NTOK + n0) * HD;
  const unsigned short* Kb = Kw + bh * NTOK * HD;
  const unsigned short* Vb = Vw + bh * HD * NTOK;

  int lane = threadIdx.x & 31, lh = lane >> 4, nn = lane & 15;

  v8f O[4] = {};                    // 64(c) x 16(n) accumulator
  float rmax = -1e30f, rsum = 0.0f;

  // Q B-fragments reused for every m-chunk (K-dim = head-dim 64 = 2 x 32)
  v16bf bq0 = load_b_bf16(Qb + 0,  HD);
  v16bf bq1 = load_b_bf16(Qb + 32, HD);

  for (int mb = 0; mb < NTOK; mb += 32) {
    // S^T tiles: rows m, cols n  (D layout: lane=col n, VGPR r -> m=8*lh+r)
    v8f s0 = {}, s1 = {};
    s0 = wmma_bf16(load_a_bf16(Kb + (long)mb * HD, HD),        bq0, s0);
    s0 = wmma_bf16(load_a_bf16(Kb + (long)mb * HD + 32, HD),   bq1, s0);
    s1 = wmma_bf16(load_a_bf16(Kb + (long)(mb + 16) * HD, HD),      bq0, s1);
    s1 = wmma_bf16(load_a_bf16(Kb + (long)(mb + 16) * HD + 32, HD), bq1, s1);

    // ---- online softmax over m for this chunk (per column n) ----
    float cmax = -1e30f;
#pragma unroll
    for (int r = 0; r < 8; ++r) cmax = fmaxf(cmax, fmaxf(s0[r], s1[r]));
    cmax = fmaxf(cmax, __shfl_xor(cmax, 16, 32));
    float nmax = fmaxf(rmax, cmax);
    float corr = __expf(rmax - nmax);

    float e0[8], e1[8], csum = 0.0f;
#pragma unroll
    for (int r = 0; r < 8; ++r) {
      e0[r] = __expf(s0[r] - nmax);
      e1[r] = __expf(s1[r] - nmax);
      csum += e0[r] + e1[r];
    }
    csum += __shfl_xor(csum, 16, 32);
    rsum = rsum * corr + csum;
    rmax = nmax;
#pragma unroll
    for (int t = 0; t < 4; ++t)
#pragma unroll
      for (int r = 0; r < 8; ++r) O[t][r] *= corr;

    // ---- stage P (bf16) into LDS: row n, m contiguous ----
    uint4 pk0, pk1;
    pk0.x = pack2(e0[0], e0[1]); pk0.y = pack2(e0[2], e0[3]);
    pk0.z = pack2(e0[4], e0[5]); pk0.w = pack2(e0[6], e0[7]);
    pk1.x = pack2(e1[0], e1[1]); pk1.y = pack2(e1[2], e1[3]);
    pk1.z = pack2(e1[4], e1[5]); pk1.w = pack2(e1[6], e1[7]);
    *reinterpret_cast<uint4*>(P + nn * 16 + 4 * lh)     = pk0;  // m = mb+8lh..+7
    *reinterpret_cast<uint4*>(P + nn * 16 + 8 + 4 * lh) = pk1;  // m = mb+16+8lh..
    __syncthreads();

    v16bf bp = load_b_lds(P);                 // P^T 32x16 as B-fragment
#pragma unroll
    for (int ct = 0; ct < 4; ++ct)
      O[ct] = wmma_bf16(load_a_bf16(Vb + (long)ct * 16 * NTOK + mb, NTOK), bp, O[ct]);
    __syncthreads();
  }

  // normalize & write Ow[b][n][c] (c contiguous per lane -> b128 stores)
  float inv = 1.0f / rsum;
  unsigned short* orow = Ow + ((long)b * NTOK + n0 + nn) * CCH + h * HD;
#pragma unroll
  for (int ct = 0; ct < 4; ++ct) {
    uint4 pk;
    pk.x = pack2(O[ct][0] * inv, O[ct][1] * inv);
    pk.y = pack2(O[ct][2] * inv, O[ct][3] * inv);
    pk.z = pack2(O[ct][4] * inv, O[ct][5] * inv);
    pk.w = pack2(O[ct][6] * inv, O[ct][7] * inv);
    *reinterpret_cast<uint4*>(orow + ct * 16 + 8 * lh) = pk;
  }
}

// ---------------------------------------------------------------------------
// Kernel 4: proj GEMM + bias + residual. out (B,C,N) f32.
// ---------------------------------------------------------------------------
__global__ void k_proj(const unsigned short* __restrict__ w,  // bf16 (512,512)
                       const float* __restrict__ bias,
                       const unsigned short* __restrict__ Ow,
                       const float* __restrict__ x, float* __restrict__ out) {
  int wid = blockIdx.x * (blockDim.x >> 5) + (threadIdx.x >> 5);
  int nt = wid & 31;           // 32 n-tiles of 32
  int ot = (wid >> 5) & 15;    // 16 o-tiles of 32
  int b  = wid / (32 * 16);
  int o0 = ot * 32, n0 = nt * 32;

  v8f acc[2][2] = {};
  const unsigned short* Bbase = Ow + ((long)b * NTOK + n0) * CCH;
#pragma unroll 4
  for (int kb = 0; kb < CCH; kb += 32) {
    v16bf a0 = load_a_bf16(w + (long)o0 * CCH + kb, CCH);
    v16bf a1 = load_a_bf16(w + (long)(o0 + 16) * CCH + kb, CCH);
    v16bf b0 = load_b_bf16(Bbase + kb, CCH);
    v16bf b1 = load_b_bf16(Bbase + 16 * CCH + kb, CCH);
    acc[0][0] = wmma_bf16(a0, b0, acc[0][0]);
    acc[0][1] = wmma_bf16(a0, b1, acc[0][1]);
    acc[1][0] = wmma_bf16(a1, b0, acc[1][0]);
    acc[1][1] = wmma_bf16(a1, b1, acc[1][1]);
  }

  int lane = threadIdx.x & 31, lh = lane >> 4, nn = lane & 15;
#pragma unroll
  for (int i = 0; i < 2; ++i) {
    int orow0 = o0 + 16 * i + 8 * lh;
#pragma unroll
    for (int j = 0; j < 2; ++j) {
      int n = n0 + 16 * j + nn;
      long base = ((long)b * CCH + orow0) * NTOK + n;
#pragma unroll
      for (int r = 0; r < 8; ++r)
        out[base + (long)r * NTOK] =
            acc[i][j][r] + bias[orow0 + r] + x[base + (long)r * NTOK];
    }
  }
}

// ---------------------------------------------------------------------------
extern "C" void kernel_launch(void* const* d_in, const int* in_sizes, int n_in,
                              void* d_out, int out_size, void* d_ws, size_t ws_size,
                              hipStream_t stream) {
  const float* x      = (const float*)d_in[0];
  const float* qkv_w  = (const float*)d_in[1];
  const float* qkv_b  = (const float*)d_in[2];
  const float* proj_w = (const float*)d_in[3];
  const float* proj_b = (const float*)d_in[4];
  float* out = (float*)d_out;

  // workspace (bf16 elements):
  //   [0,R)    xT, later reused as attention output Ow
  //   [R,2R)   Qw   [2R,3R) Kw   [3R,4R) Vw
  //   [4R,..)  qkv_w bf16 (786432) ; then proj_w bf16 (262144)
  const size_t R = (size_t)BATCH * NTOK * CCH;   // 8,388,608 elements
  unsigned short* xT = (unsigned short*)d_ws;    // aliased with Ow
  unsigned short* Qw = xT + R;
  unsigned short* Kw = xT + 2 * R;
  unsigned short* Vw = xT + 3 * R;
  unsigned short* Wq = xT + 4 * R;               // 1536*512
  unsigned short* Wp = Wq + (size_t)1536 * 512;  // 512*512
  unsigned short* Ow = xT;

  // 0) weight f32->bf16: (1536*512)/4/256 = 768 blocks ; (512*512)/4/256 = 256
  k_cvt<<<768, 256, 0, stream>>>(qkv_w, Wq);
  k_cvt<<<256, 256, 0, stream>>>(proj_w, Wp);
  // 1) LDS-tiled transpose + f32->bf16: 2048 blocks
  k_xpose<<<2048, 256, 0, stream>>>(x, xT);
  // 2) qkv gemm: 16b * 48 * 32 = 24576 waves, 4 waves/block
  k_qkv<<<6144, 128, 0, stream>>>(Wq, qkv_b, xT, Qw, Kw, Vw);
  // 3) flash attention: 16*8*64 = 8192 waves
  k_attn<<<2048, 128, 0, stream>>>(Qw, Kw, Vw, Ow);
  // 4) proj + residual: 16*16*32 = 8192 waves
  k_proj<<<2048, 128, 0, stream>>>(proj_w ? Wp : Wp, proj_b, Ow, x, out);
}
